// RNN_51608327029090
// MI455X (gfx1250) — matrix operand order repaired and checked by reading, not compiled, not run on hardware
//
#include <hip/hip_runtime.h>

// ---------------------------------------------------------------------------
// RNN (2-layer tanh) for MI455X / gfx1250, bf16 WMMA path.
// S=128, B=64, E=H=1024, V=10000, L=2
// ---------------------------------------------------------------------------

#define RNN_S 128
#define RNN_B 64
#define RNN_E 1024
#define RNN_H 1024
#define RNN_V 10000

typedef __attribute__((ext_vector_type(16))) __bf16 v16bf;
typedef __attribute__((ext_vector_type(8)))  float  v8f;

union FragBF16 {
    uint4  u[2];   // 32 bytes
    v16bf  v;
};

// fp32 -> bf16, round-to-nearest-even
__device__ __forceinline__ unsigned short f2bf(float f) {
    unsigned int u = __float_as_uint(f);
    u += 0x7FFFu + ((u >> 16) & 1u);
    return (unsigned short)(u >> 16);
}

// ---------------------------------------------------------------------------
// Utility kernels
// ---------------------------------------------------------------------------
__global__ void cvt_f32_to_bf16(const float* __restrict__ in,
                                unsigned short* __restrict__ out, long n) {
    long i = blockIdx.x * (long)blockDim.x + threadIdx.x;
    long stride = (long)gridDim.x * blockDim.x;
    for (; i < n; i += stride) out[i] = f2bf(in[i]);
}

// copy fp32 -> (fp32, bf16) pair (hidden-state init)
__global__ void cvt_dual(const float* __restrict__ in,
                         float* __restrict__ outF,
                         unsigned short* __restrict__ outB, int n) {
    int i = blockIdx.x * blockDim.x + threadIdx.x;
    int stride = gridDim.x * blockDim.x;
    for (; i < n; i += stride) {
        float v = in[i];
        outF[i] = v;
        outB[i] = f2bf(v);
    }
}

__global__ void copy_f32(const float* __restrict__ in,
                         float* __restrict__ out, int n) {
    int i = blockIdx.x * blockDim.x + threadIdx.x;
    int stride = gridDim.x * blockDim.x;
    for (; i < n; i += stride) out[i] = in[i];
}

// Gather embedding rows (indexed) into contiguous bf16 A matrix.
__global__ void gather_emb(const int* __restrict__ idx,
                           const float* __restrict__ emb,
                           unsigned short* __restrict__ out, int E) {
    int row = blockIdx.x;
    const float* src = emb + (long)idx[row] * E;
    unsigned short* dst = out + (long)row * E;
    for (int e = threadIdx.x; e < E; e += blockDim.x) dst[e] = f2bf(src[e]);
}

// ---------------------------------------------------------------------------
// WMMA GEMM:  C[M,N] = sum_pass A_pass[M,K](bf16) * W_pass[N,K](bf16)^T
//             (+ addin[M,N] fp32) (+ bias[N]) (optional tanh)
// One wave per block; each wave computes a (16*MT) x 64 tile.
// Inner loop is branch-free: OOB n-tiles are clamped to N-16 for loads
// (wave-uniform; duplicated compute, stores guarded in epilogue), so EXEC
// stays all-ones as WMMA requires. K processed 64 elements/iteration with
// two fragment sets so loads of set 1 overlap WMMAs of set 0.
// Requires M % (16*MT) == 0, K % 64 == 0, N % 16 == 0.
// ---------------------------------------------------------------------------
template<int MT, bool HAS_A2, bool HAS_ADDIN, bool HAS_BIAS, bool DO_TANH,
         bool OUT_F, bool OUT_B>
__global__ void __launch_bounds__(32) wmma_gemm_bf16(
    const unsigned short* __restrict__ A1, const unsigned short* __restrict__ W1,
    const unsigned short* __restrict__ A2, const unsigned short* __restrict__ W2,
    const float* __restrict__ addin, const float* __restrict__ bias,
    float* __restrict__ outF, unsigned short* __restrict__ outB,
    int N, int K)
{
    const int lane  = threadIdx.x;     // 0..31 (wave32)
    const int lmod  = lane & 15;
    const int lhalf = lane >> 4;       // 0 or 1
    const int m0 = blockIdx.y * (16 * MT);
    const int nb = blockIdx.x * 64;

    // Clamped n-tile bases (uniform).
    int n0c[4];
    #pragma unroll
    for (int j = 0; j < 4; ++j) {
        int n0 = nb + j * 16;
        n0c[j] = (n0 + 16 <= N) ? n0 : (N - 16);
    }

    const v8f vzero = {0.f, 0.f, 0.f, 0.f, 0.f, 0.f, 0.f, 0.f};
    v8f acc[MT][4];
    #pragma unroll
    for (int mt = 0; mt < MT; ++mt)
        #pragma unroll
        for (int j = 0; j < 4; ++j) acc[mt][j] = vzero;

    const int npass = HAS_A2 ? 2 : 1;
    #pragma unroll
    for (int pass = 0; pass < npass; ++pass) {
        const unsigned short* A = (HAS_A2 && pass) ? A2 : A1;
        const unsigned short* W = (HAS_A2 && pass) ? W2 : W1;

        // Per-lane base pointers.
        // A-operand (16x32 bf16): lane<16 -> K {0..7, 16..23}; lane>=16 -> {8..15, 24..31}
        const unsigned short* arow[MT];
        #pragma unroll
        for (int mt = 0; mt < MT; ++mt)
            arow[mt] = A + (long)(m0 + mt * 16 + lmod) * K + lhalf * 8;
        // B-operand: lane holds W row (n0 + lane%16); lane<16 -> K {0..7, 16..23};
        // lane>=16 -> {8..15, 24..31}.
        const unsigned short* wrow[4];
        #pragma unroll
        for (int j = 0; j < 4; ++j)
            wrow[j] = W + (long)(n0c[j] + lmod) * K + lhalf * 8;

        for (int kk = 0; kk < K; kk += 64) {
            FragBF16 a0[MT], a1[MT], b0[4], b1[4];
            // Issue ALL loads for both 32-wide K chunks first.
            #pragma unroll
            for (int mt = 0; mt < MT; ++mt) {
                const unsigned short* ap = arow[mt] + kk;
                a0[mt].u[0] = *reinterpret_cast<const uint4*>(ap);
                a0[mt].u[1] = *reinterpret_cast<const uint4*>(ap + 16);
                a1[mt].u[0] = *reinterpret_cast<const uint4*>(ap + 32);
                a1[mt].u[1] = *reinterpret_cast<const uint4*>(ap + 48);
            }
            #pragma unroll
            for (int j = 0; j < 4; ++j) {
                const unsigned short* wp = wrow[j] + kk;
                b0[j].u[0] = *reinterpret_cast<const uint4*>(wp);
                b0[j].u[1] = *reinterpret_cast<const uint4*>(wp + 16);
                b1[j].u[0] = *reinterpret_cast<const uint4*>(wp + 32);
                b1[j].u[1] = *reinterpret_cast<const uint4*>(wp + 48);
            }
            // Chunk 0 WMMAs (can start once its loads land; chunk 1 loads
            // still in flight).
            #pragma unroll
            for (int mt = 0; mt < MT; ++mt)
                #pragma unroll
                for (int j = 0; j < 4; ++j)
                    acc[mt][j] = __builtin_amdgcn_wmma_f32_16x16x32_bf16(
                        false, a0[mt].v, false, b0[j].v, (short)0,
                        acc[mt][j], false, false);
            // Chunk 1 WMMAs.
            #pragma unroll
            for (int mt = 0; mt < MT; ++mt)
                #pragma unroll
                for (int j = 0; j < 4; ++j)
                    acc[mt][j] = __builtin_amdgcn_wmma_f32_16x16x32_bf16(
                        false, a1[mt].v, false, b1[j].v, (short)0,
                        acc[mt][j], false, false);
        }
    }

    // Epilogue. C/D layout: acc[r] -> C[m0 + mt*16 + 8*lhalf + r][n0 + lmod]
    #pragma unroll
    for (int j = 0; j < 4; ++j) {
        int n0 = nb + j * 16;
        if (n0 + 16 > N) continue;                        // uniform
        int n = n0 + lmod;
        float bv = HAS_BIAS ? bias[n] : 0.0f;
        #pragma unroll
        for (int mt = 0; mt < MT; ++mt) {
            #pragma unroll
            for (int r = 0; r < 8; ++r) {
                int m = m0 + mt * 16 + 8 * lhalf + r;
                float v = acc[mt][j][r] + bv;
                if (HAS_ADDIN) v += addin[(long)m * N + n];
                if (DO_TANH)   v = tanhf(v);
                if (OUT_F)     outF[(long)m * N + n] = v;
                if (OUT_B)     outB[(long)m * N + n] = f2bf(v);
            }
        }
    }
}

// ---------------------------------------------------------------------------
// Host launch
// ---------------------------------------------------------------------------
extern "C" void kernel_launch(void* const* d_in, const int* in_sizes, int n_in,
                              void* d_out, int out_size, void* d_ws, size_t ws_size,
                              hipStream_t stream) {
    (void)in_sizes; (void)n_in; (void)out_size; (void)ws_size;
    const int S = RNN_S, B = RNN_B, E = RNN_E, H = RNN_H, V = RNN_V;

    const int*   inputs  = (const int*)  d_in[0];   // (S,B)
    const float* hidden0 = (const float*)d_in[1];   // (L,B,H)
    const float* emb     = (const float*)d_in[2];   // (V,E)
    const float* Wx0     = (const float*)d_in[3];   // (H,E)
    const float* Wx1     = (const float*)d_in[4];   // (1,H,H)
    const float* Wh      = (const float*)d_in[5];   // (2,H,H)
    const float* bh      = (const float*)d_in[6];   // (2,H)
    const float* Wout    = (const float*)d_in[7];   // (V,H)
    const float* bout    = (const float*)d_in[8];   // (V,)
    float* out = (float*)d_out;                     // logits (S,B,V) ++ h_final (2,B,H)

    // Workspace carve-up (256B aligned)
    char* ws = (char*)d_ws;
    size_t off = 0;
    auto alloc = [&](size_t bytes) -> char* {
        char* p = ws + off;
        off += (bytes + 255) & ~(size_t)255;
        return p;
    };
    unsigned short* A0     = (unsigned short*)alloc((size_t)S * B * E * 2);
    unsigned short* Wx0b   = (unsigned short*)alloc((size_t)H * E * 2);
    unsigned short* Wx1b   = (unsigned short*)alloc((size_t)H * H * 2);
    unsigned short* Wh0b   = (unsigned short*)alloc((size_t)H * H * 2);
    unsigned short* Wh1b   = (unsigned short*)alloc((size_t)H * H * 2);
    unsigned short* Woutb  = (unsigned short*)alloc((size_t)V * H * 2);
    float*          x0f    = (float*)        alloc((size_t)S * B * H * 4);
    unsigned short* d1seq  = (unsigned short*)alloc((size_t)S * B * H * 2);
    float*          h0f    = (float*)        alloc((size_t)B * H * 4);
    float*          h1f    = (float*)        alloc((size_t)B * H * 4);
    unsigned short* h0b0   = (unsigned short*)alloc((size_t)B * H * 2);
    unsigned short* h0b1   = (unsigned short*)alloc((size_t)B * H * 2);
    unsigned short* h1binit= (unsigned short*)alloc((size_t)B * H * 2);
    unsigned short* h0b[2] = {h0b0, h0b1};

    const int T = 256;

    // 1) weight conversions + embedding gather + hidden init
    cvt_f32_to_bf16<<<2048, T, 0, stream>>>(Wx0,              Wx0b, (long)H * E);
    cvt_f32_to_bf16<<<2048, T, 0, stream>>>(Wx1,              Wx1b, (long)H * H);
    cvt_f32_to_bf16<<<2048, T, 0, stream>>>(Wh,               Wh0b, (long)H * H);
    cvt_f32_to_bf16<<<2048, T, 0, stream>>>(Wh + (long)H * H, Wh1b, (long)H * H);
    cvt_f32_to_bf16<<<4096, T, 0, stream>>>(Wout,             Woutb, (long)V * H);
    gather_emb<<<S * B, T, 0, stream>>>(inputs, emb, A0, E);
    cvt_dual<<<256, T, 0, stream>>>(hidden0,               h0f, h0b[0], B * H);
    cvt_dual<<<256, T, 0, stream>>>(hidden0 + (long)B * H, h1f, h1binit, B * H);

    // 2) x0 = emb_seq @ Wx0^T   (M=S*B, N=H, K=E), fp32 result, MT=2
    {
        dim3 grid(H / 64, (S * B) / 32);
        wmma_gemm_bf16<2, false, false, false, false, true, false>
            <<<grid, 32, 0, stream>>>(
                A0, Wx0b, nullptr, nullptr, nullptr, nullptr,
                x0f, nullptr, H, E);
    }

    // 3) sequential recurrence (weights L2-resident: 3 x 2MB bf16); MT=1 for
    //    max wave parallelism in the latency-bound small GEMMs.
    for (int s = 0; s < S; ++s) {
        dim3 grid(H / 64, B / 16);
        unsigned short* h0in  = h0b[s & 1];
        unsigned short* h0out = h0b[(s + 1) & 1];
        // layer 0: h0 = tanh(x0_t + h0 @ Wh0^T + bh0)
        wmma_gemm_bf16<1, false, true, true, true, true, true>
            <<<grid, 32, 0, stream>>>(
                h0in, Wh0b, nullptr, nullptr,
                x0f + (size_t)s * B * H, bh,
                h0f, h0out, H, H);
        // layer 1: d = tanh(h0_new @ Wx1^T + h1 @ Wh1^T + bh1)
        const unsigned short* h1in =
            (s == 0) ? h1binit : (d1seq + (size_t)(s - 1) * B * H);
        wmma_gemm_bf16<1, true, false, true, true, true, true>
            <<<grid, 32, 0, stream>>>(
                h0out, Wx1b, h1in, Wh1b,
                nullptr, bh + H,
                h1f, d1seq + (size_t)s * B * H, H, H);
    }

    // 4) logits = d1seq @ Wout^T + bout   (M=S*B, N=V, K=H), MT=2
    {
        dim3 grid((V + 63) / 64, (S * B) / 32);
        wmma_gemm_bf16<2, false, false, true, false, true, false>
            <<<grid, 32, 0, stream>>>(
                d1seq, Woutb, nullptr, nullptr, nullptr, bout,
                out, nullptr, V, H);
    }

    // 5) h_final = [h0_last, h1_last]
    copy_f32<<<256, T, 0, stream>>>(h0f, out + (size_t)S * B * V, B * H);
    copy_f32<<<256, T, 0, stream>>>(h1f, out + (size_t)S * B * V + (size_t)B * H, B * H);
}
